// AdaDepression_47931835023415
// MI455X (gfx1250) — matrix operand order repaired and checked by reading, not compiled, not run on hardware
//
#include <hip/hip_runtime.h>

// MI455X / gfx1250: wave32, WMMA 16x16x32 f16 -> f32 accum.
typedef __attribute__((ext_vector_type(16))) _Float16 v16h;
typedef __attribute__((ext_vector_type(8)))  _Float16 v8h;
typedef __attribute__((ext_vector_type(8)))  float    v8f;

#define WMMA16(a, b, c) \
  __builtin_amdgcn_wmma_f32_16x16x32_f16(false, (a), false, (b), (short)0, (c), false, false)

#define B_ROWS 8192
#define N_EMB  1024
#define D_IN   384
#define H_DIM  64
#define E_EXP  8

// ---------------------------------------------------------------------------
// Bulk f32 -> f16 conversion (vectorized, one-time).
// ---------------------------------------------------------------------------
__global__ void f32_to_f16_kernel(const float* __restrict__ in,
                                  _Float16* __restrict__ out, int n) {
  const int i = (blockIdx.x * blockDim.x + threadIdx.x) * 8;
  if (i >= n) return;
#pragma unroll
  for (int j = 0; j < 8; ++j) out[i + j] = (_Float16)in[i + j];
}

// ---------------------------------------------------------------------------
// Weight transpose to f16: Wt[e][n][k] = W[e][k][n]   (k-contiguous B loads)
// grid = (E, 64), block = 128.
// ---------------------------------------------------------------------------
__global__ void w_transpose_f16_kernel(const float* __restrict__ W,
                                       _Float16* __restrict__ Wt, int d_in) {
  const int e = blockIdx.x, n = blockIdx.y;
  const float* src = W + (size_t)e * d_in * 64 + n;
  _Float16* dst = Wt + ((size_t)e * 64 + n) * d_in;
  for (int k = threadIdx.x; k < d_in; k += blockDim.x)
    dst[k] = (_Float16)src[(size_t)k * 64];
}

// ---------------------------------------------------------------------------
// Gate: logits = x @ gate_w + b ; softmax probs ; top-2 ; gate weights ;
// deterministic per-block aux partials: [probs_sum x8, mask_sum x8].
// ---------------------------------------------------------------------------
__global__ void gate_topk_kernel(const float* __restrict__ X, int d_in,
                                 const float* __restrict__ GW,
                                 const float* __restrict__ GB,
                                 int* __restrict__ topk, float* __restrict__ gatew,
                                 float* __restrict__ part) {
  __shared__ float swred[8][16];
  const int tid = threadIdx.x;
  const int b = blockIdx.x * blockDim.x + tid;

  float lg[8];
#pragma unroll
  for (int e = 0; e < 8; ++e) lg[e] = GB[e];
  const float* xr = X + (size_t)b * d_in;
  for (int d = 0; d < d_in; ++d) {
    const float xv = xr[d];
    const float* g = GW + (size_t)d * 8;
#pragma unroll
    for (int e = 0; e < 8; ++e) lg[e] = fmaf(xv, g[e], lg[e]);
  }
  float mx = lg[0];
#pragma unroll
  for (int e = 1; e < 8; ++e) mx = fmaxf(mx, lg[e]);
  float pr[8], se = 0.f;
#pragma unroll
  for (int e = 0; e < 8; ++e) { pr[e] = expf(lg[e] - mx); se += pr[e]; }
#pragma unroll
  for (int e = 0; e < 8; ++e) pr[e] /= se;
  int i0 = 0;
#pragma unroll
  for (int e = 1; e < 8; ++e) if (lg[e] > lg[i0]) i0 = e;
  int i1 = (i0 == 0) ? 1 : 0;
#pragma unroll
  for (int e = 0; e < 8; ++e) if (e != i1 && e != i0 && lg[e] > lg[i1]) i1 = e;
  const float ex1 = expf(lg[i1] - lg[i0]);
  const float w0 = 1.f / (1.f + ex1);
  const float w1 = ex1 / (1.f + ex1);
  topk[b * 2 + 0] = i0; topk[b * 2 + 1] = i1;
  gatew[b * 2 + 0] = w0; gatew[b * 2 + 1] = w1;

  float mk[8];
#pragma unroll
  for (int e = 0; e < 8; ++e) mk[e] = (e == i0 || e == i1) ? 1.f : 0.f;

  const int wave = tid >> 5, lane = tid & 31;
#pragma unroll
  for (int s = 0; s < 16; ++s) {
    float v = (s < 8) ? pr[s] : mk[s - 8];
#pragma unroll
    for (int msk = 1; msk < 32; msk <<= 1) v += __shfl_xor(v, msk, 32);
    if (lane == 0) swred[wave][s] = v;
  }
  __syncthreads();
  if (tid < 16) {
    float v = 0.f;
#pragma unroll
    for (int w = 0; w < 8; ++w) v += swred[w][tid];
    part[blockIdx.x * 16 + tid] = v;
  }
}

// ---------------------------------------------------------------------------
// proj_norm: out[row][e][0:64] = normalize(X[row,:] @ W[e]^T + bias[e]) as f16.
// X is f16 [nrows][d_in]; Wt is f16 [E][64][d_in] (k-contiguous).
// One wave per 16-row tile; all loads are contiguous 16/32-byte vectors.
// grid = (nrows/128, E), block = 256 (8 waves).
// ---------------------------------------------------------------------------
__global__ void proj_norm_kernel(const _Float16* __restrict__ X, int d_in, int nrows,
                                 const _Float16* __restrict__ Wt,
                                 const float* __restrict__ Bv,
                                 _Float16* __restrict__ out) {
  const int lane = threadIdx.x & 31;
  const int wave = threadIdx.x >> 5;
  const int e = blockIdx.y;
  const int row_base = (blockIdx.x * 8 + wave) * 16;
  if (row_base >= nrows) return;

  const int m = lane & 15;                  // A row / B col within tile
  const int khalf8 = ((lane >> 4) & 1) << 3;
  const int kh16 = ((lane >> 4) & 1) << 4;

  v8f acc[4] = {v8f{}, v8f{}, v8f{}, v8f{}};
  const _Float16* xrow = X + (size_t)(row_base + m) * d_in;
  const _Float16* wbase = Wt + (size_t)e * 64 * d_in + (size_t)m * d_in;

  for (int k0 = 0; k0 < d_in; k0 += 32) {
    // A fragment: two contiguous 8-half chunks selected by lane half
    const v8h alo = *(const v8h*)(xrow + k0 + khalf8);
    const v8h ahi = *(const v8h*)(xrow + k0 + 16 + khalf8);
    v16h a;
#pragma unroll
    for (int j = 0; j < 8; ++j) { a[j] = alo[j]; a[8 + j] = ahi[j]; }
#pragma unroll
    for (int t = 0; t < 4; ++t) {
      // B fragment: two 8-half chunks of Wt[e][t*16+m][k0+khalf8 ...] and [k0+16+khalf8 ...]
      const _Float16* wp = wbase + (size_t)(t * 16) * d_in + k0 + khalf8;
      const v8h blo = *(const v8h*)wp;
      const v8h bhi = *(const v8h*)(wp + 16);
      v16h bf;
#pragma unroll
      for (int j = 0; j < 8; ++j) { bf[j] = blo[j]; bf[8 + j] = bhi[j]; }
      acc[t] = WMMA16(a, bf, acc[t]);
    }
  }
  // bias add: element i of acc[t] is (row = i + 8*(lane>=16), col = t*16 + m)
  const float* be = Bv + e * 64 + m;
#pragma unroll
  for (int t = 0; t < 4; ++t) {
    const float bb = be[t * 16];
#pragma unroll
    for (int i = 0; i < 8; ++i) acc[t][i] += bb;
  }
#pragma unroll
  for (int i = 0; i < 8; ++i) {
    float ss = acc[0][i] * acc[0][i] + acc[1][i] * acc[1][i] +
               acc[2][i] * acc[2][i] + acc[3][i] * acc[3][i];
    ss += __shfl_xor(ss, 1, 32);
    ss += __shfl_xor(ss, 2, 32);
    ss += __shfl_xor(ss, 4, 32);
    ss += __shfl_xor(ss, 8, 32);
    const float inv = rsqrtf(ss);
    const int row = row_base + i + khalf8;
    _Float16* op = out + ((size_t)row * 8 + e) * 64 + m;
#pragma unroll
    for (int t = 0; t < 4; ++t) op[t * 16] = (_Float16)(acc[t][i] * inv);
  }
}

// ---------------------------------------------------------------------------
// score_sample: for a 16-row tile, loop all 8 experts:
//   S = u_tile[16x64] x v[e]^T[64x1024] via WMMA, softmax over N,
//   p[row,:] += wtab[row][e] * softmax(S[row,:])   (wtab = top2 gate weights)
// then per-row cumsum over N, find first cum > rand, emit idx + log p[idx].
// grid = B/16, block = 256 (8 waves x 128 cols), dyn LDS ~65KB.
// ---------------------------------------------------------------------------
__global__ void score_sample_kernel(const _Float16* __restrict__ U,
                                    const _Float16* __restrict__ V,
                                    const int* __restrict__ topk,
                                    const float* __restrict__ gatew,
                                    const float* __restrict__ rnd,
                                    int* __restrict__ selOut,
                                    float* __restrict__ outIdx,
                                    float* __restrict__ outLogp) {
  extern __shared__ float smem[];
  float* pbuf = smem;                 // 16*1024
  float* wtab = pbuf + 16 * 1024;     // 16*8
  float* redw = wtab + 128;           // 16*8
  float* rowv = redw + 128;           // 16

  const int tid = threadIdx.x;
  const int lane = tid & 31, wave = tid >> 5;
  const int row_base = blockIdx.x * 16;
  const int m = lane & 15;
  const int half8 = ((lane >> 4) & 1) << 3;
  const int kh16 = ((lane >> 4) & 1) << 4;

  for (int i = tid; i < 16 * 1024; i += 256) pbuf[i] = 0.f;
  if (tid < 16) {
    const int b = row_base + tid;
#pragma unroll
    for (int e = 0; e < 8; ++e) wtab[tid * 8 + e] = 0.f;
    wtab[tid * 8 + topk[b * 2 + 0]] += gatew[b * 2 + 0];
    wtab[tid * 8 + topk[b * 2 + 1]] += gatew[b * 2 + 1];
  }
  __syncthreads();

  for (int e = 0; e < 8; ++e) {
    // A fragments (K=64 -> two k-steps of 32) from u[row][e][:], vector loads
    const _Float16* up = U + ((size_t)(row_base + m) * 8 + e) * 64;
    const v8h a0l = *(const v8h*)(up + half8);
    const v8h a0h = *(const v8h*)(up + 16 + half8);
    const v8h a1l = *(const v8h*)(up + 32 + half8);
    const v8h a1h = *(const v8h*)(up + 48 + half8);
    v16h a0, a1;
#pragma unroll
    for (int j = 0; j < 8; ++j) {
      a0[j] = a0l[j]; a0[8 + j] = a0h[j];
      a1[j] = a1l[j]; a1[8 + j] = a1h[j];
    }
    v8f s[8];
#pragma unroll
    for (int nt = 0; nt < 8; ++nt) {
      const int n = wave * 128 + nt * 16 + m;
      const _Float16* vp = V + ((size_t)n * 8 + e) * 64 + half8;
      __builtin_prefetch(vp + 512, 0, 1);  // global_prefetch of upcoming v rows
      const v8h b0l = *(const v8h*)vp;
      const v8h b0h = *(const v8h*)(vp + 16);
      const v8h b1l = *(const v8h*)(vp + 32);
      const v8h b1h = *(const v8h*)(vp + 48);
      v16h b0, b1;
#pragma unroll
      for (int j = 0; j < 8; ++j) {
        b0[j] = b0l[j]; b0[8 + j] = b0h[j];
        b1[j] = b1l[j]; b1[8 + j] = b1h[j];
      }
      v8f c = {};
      c = WMMA16(a0, b0, c);
      c = WMMA16(a1, b1, c);
      s[nt] = c;
    }
    // row max: within-wave (128 cols) then cross-wave via LDS
#pragma unroll
    for (int i = 0; i < 8; ++i) {
      float mx = s[0][i];
#pragma unroll
      for (int nt = 1; nt < 8; ++nt) mx = fmaxf(mx, s[nt][i]);
      mx = fmaxf(mx, __shfl_xor(mx, 1, 32));
      mx = fmaxf(mx, __shfl_xor(mx, 2, 32));
      mx = fmaxf(mx, __shfl_xor(mx, 4, 32));
      mx = fmaxf(mx, __shfl_xor(mx, 8, 32));
      if (m == 0) redw[(i + half8) * 8 + wave] = mx;
    }
    __syncthreads();
    if (tid < 16) {
      float v0 = redw[tid * 8];
#pragma unroll
      for (int w = 1; w < 8; ++w) v0 = fmaxf(v0, redw[tid * 8 + w]);
      rowv[tid] = v0;
    }
    __syncthreads();
    // exp + row sum
#pragma unroll
    for (int i = 0; i < 8; ++i) {
      const float rm = rowv[i + half8];
      float sm = 0.f;
#pragma unroll
      for (int nt = 0; nt < 8; ++nt) { s[nt][i] = expf(s[nt][i] - rm); sm += s[nt][i]; }
      sm += __shfl_xor(sm, 1, 32);
      sm += __shfl_xor(sm, 2, 32);
      sm += __shfl_xor(sm, 4, 32);
      sm += __shfl_xor(sm, 8, 32);
      if (m == 0) redw[(i + half8) * 8 + wave] = sm;
    }
    __syncthreads();
    if (tid < 16) {
      float v0 = 0.f;
#pragma unroll
      for (int w = 0; w < 8; ++w) v0 += redw[tid * 8 + w];
      rowv[tid] = v0;
    }
    __syncthreads();
    // accumulate gated softmax into p
#pragma unroll
    for (int i = 0; i < 8; ++i) {
      const int r = i + half8;
      const float scale = wtab[r * 8 + e] / rowv[r];
#pragma unroll
      for (int nt = 0; nt < 8; ++nt)
        pbuf[r * 1024 + wave * 128 + nt * 16 + m] += s[nt][i] * scale;
    }
    __syncthreads();
  }

  // cumsum + inverse-CDF sampling: wave handles rows {wave, wave+8}
  for (int r = wave; r < 16; r += 8) {
    const int b = row_base + r;
    const float rv = rnd[b];
    float run = 0.f;
    int idx = -1;
    float pv = 0.f;
    for (int c0 = 0; c0 < 1024; c0 += 32) {
      float v = pbuf[r * 1024 + c0 + lane];
#pragma unroll
      for (int d = 1; d < 32; d <<= 1) {
        const float t = __shfl_up(v, d, 32);
        if (lane >= d) v += t;
      }
      const float cum = run + v;
      const unsigned long long bm = __ballot(cum > rv);
      if (idx < 0 && bm != 0ull) {
        const int first = __ffsll((unsigned long long)bm) - 1;
        idx = c0 + first;
        pv = pbuf[r * 1024 + idx];
      }
      run += __shfl(v, 31, 32);
    }
    if (idx < 0) { idx = 0; pv = pbuf[r * 1024]; }  // argmax of all-false -> 0
    if (lane == 0) {
      selOut[b] = idx;
      outIdx[b] = (float)idx;
      outLogp[b] = logf(pv);
    }
  }
}

// ---------------------------------------------------------------------------
// concat: xc[b] = [x[b], emb[sel[b]]]  (writes f32 for gate + f16 for WMMA)
// ---------------------------------------------------------------------------
__global__ void concat_kernel(const float* __restrict__ X,
                              const float* __restrict__ EMB,
                              const int* __restrict__ sel,
                              float* __restrict__ XC,
                              _Float16* __restrict__ XCh) {
  const int b = blockIdx.x;
  const float* xr = X + (size_t)b * 384;
  const float* er = EMB + (size_t)sel[b] * 384;
  float* o = XC + (size_t)b * 768;
  _Float16* oh = XCh + (size_t)b * 768;
  for (int j = threadIdx.x; j < 384; j += blockDim.x) {
    const float a = xr[j], c = er[j];
    o[j] = a;           o[384 + j] = c;
    oh[j] = (_Float16)a; oh[384 + j] = (_Float16)c;
  }
}

// ---------------------------------------------------------------------------
// aux: deterministic final reduction of per-block partials from both stages.
// part layout: [stage][32 blocks][16 slots]
// ---------------------------------------------------------------------------
__global__ void aux_finalize_kernel(const float* __restrict__ part,
                                    float* __restrict__ out_aux) {
  float aux = 0.f;
  for (int s = 0; s < 2; ++s) {
    float pm[8], mk[8];
    for (int e = 0; e < 8; ++e) { pm[e] = 0.f; mk[e] = 0.f; }
    for (int blk = 0; blk < 32; ++blk) {
      const float* p = part + (s * 32 + blk) * 16;
      for (int e = 0; e < 8; ++e) { pm[e] += p[e]; mk[e] += p[8 + e]; }
    }
    float dot = 0.f;
    for (int e = 0; e < 8; ++e)
      dot += (pm[e] / (float)B_ROWS) * (mk[e] / (float)B_ROWS);
    aux += 8.f * dot * 0.05f;
  }
  *out_aux = aux;
}

// ---------------------------------------------------------------------------
extern "C" void kernel_launch(void* const* d_in, const int* in_sizes, int n_in,
                              void* d_out, int out_size, void* d_ws, size_t ws_size,
                              hipStream_t stream) {
  (void)in_sizes; (void)n_in; (void)out_size; (void)ws_size;
  const float* x        = (const float*)d_in[0];
  const float* rea_emb  = (const float*)d_in[1];
  const float* llm_emb  = (const float*)d_in[2];
  const float* rand_rea = (const float*)d_in[3];
  const float* rand_llm = (const float*)d_in[4];
  const float* g1w = (const float*)d_in[5];
  const float* g1b = (const float*)d_in[6];
  const float* g2w = (const float*)d_in[7];
  const float* g2b = (const float*)d_in[8];
  const float* Ur1 = (const float*)d_in[9];
  const float* Ub1 = (const float*)d_in[10];
  const float* Vr1 = (const float*)d_in[11];
  const float* Vb1 = (const float*)d_in[12];
  const float* Ur2 = (const float*)d_in[13];
  const float* Ub2 = (const float*)d_in[14];
  const float* Vr2 = (const float*)d_in[15];
  const float* Vb2 = (const float*)d_in[16];

  float* out = (float*)d_out;
  float* out_rea_idx  = out;
  float* out_rea_logp = out + B_ROWS;
  float* out_llm_idx  = out + 2 * B_ROWS;
  float* out_llm_logp = out + 3 * B_ROWS;
  float* out_aux      = out + 4 * B_ROWS;

  // workspace carve-out (256B aligned slices)
  char* w = (char*)d_ws;
  auto wsa = [&](size_t bytes) {
    char* p = w;
    w += (bytes + 255) & ~(size_t)255;
    return p;
  };
  int*      topk1 = (int*)wsa(B_ROWS * 2 * sizeof(int));
  float*    gw1   = (float*)wsa(B_ROWS * 2 * sizeof(float));
  int*      topk2 = (int*)wsa(B_ROWS * 2 * sizeof(int));
  float*    gw2   = (float*)wsa(B_ROWS * 2 * sizeof(float));
  int*      sel1  = (int*)wsa(B_ROWS * sizeof(int));
  int*      sel2  = (int*)wsa(B_ROWS * sizeof(int));
  _Float16* u1    = (_Float16*)wsa((size_t)B_ROWS * 8 * 64 * sizeof(_Float16));
  _Float16* v1    = (_Float16*)wsa((size_t)N_EMB * 8 * 64 * sizeof(_Float16));
  _Float16* u2    = (_Float16*)wsa((size_t)B_ROWS * 8 * 64 * sizeof(_Float16));
  _Float16* v2    = (_Float16*)wsa((size_t)N_EMB * 8 * 64 * sizeof(_Float16));
  float*    xc    = (float*)wsa((size_t)B_ROWS * 768 * sizeof(float));
  float*    part  = (float*)wsa(2 * 32 * 16 * sizeof(float));
  // f16 staging: inputs + transposed weights
  _Float16* x_h    = (_Float16*)wsa((size_t)B_ROWS * 384 * sizeof(_Float16));
  _Float16* remb_h = (_Float16*)wsa((size_t)N_EMB * 384 * sizeof(_Float16));
  _Float16* lemb_h = (_Float16*)wsa((size_t)N_EMB * 384 * sizeof(_Float16));
  _Float16* xc_h   = (_Float16*)wsa((size_t)B_ROWS * 768 * sizeof(_Float16));
  _Float16* Ur1t   = (_Float16*)wsa((size_t)E_EXP * 64 * 384 * sizeof(_Float16));
  _Float16* Vr1t   = (_Float16*)wsa((size_t)E_EXP * 64 * 384 * sizeof(_Float16));
  _Float16* Ur2t   = (_Float16*)wsa((size_t)E_EXP * 64 * 768 * sizeof(_Float16));
  _Float16* Vr2t   = (_Float16*)wsa((size_t)E_EXP * 64 * 384 * sizeof(_Float16));

  const size_t score_lds = (16 * 1024 + 128 + 128 + 16) * sizeof(float);

  // ---- one-time f16 staging (independent; issued up front) ----
  f32_to_f16_kernel<<<(B_ROWS * 384 / 8 + 255) / 256, 256, 0, stream>>>(
      x, x_h, B_ROWS * 384);
  f32_to_f16_kernel<<<(N_EMB * 384 / 8 + 255) / 256, 256, 0, stream>>>(
      rea_emb, remb_h, N_EMB * 384);
  f32_to_f16_kernel<<<(N_EMB * 384 / 8 + 255) / 256, 256, 0, stream>>>(
      llm_emb, lemb_h, N_EMB * 384);
  w_transpose_f16_kernel<<<dim3(E_EXP, 64), 128, 0, stream>>>(Ur1, Ur1t, 384);
  w_transpose_f16_kernel<<<dim3(E_EXP, 64), 128, 0, stream>>>(Vr1, Vr1t, 384);
  w_transpose_f16_kernel<<<dim3(E_EXP, 64), 128, 0, stream>>>(Ur2, Ur2t, 768);
  w_transpose_f16_kernel<<<dim3(E_EXP, 64), 128, 0, stream>>>(Vr2, Vr2t, 384);

  // ---- stage 1: reasoning allocation ----
  gate_topk_kernel<<<B_ROWS / 256, 256, 0, stream>>>(
      x, D_IN, g1w, g1b, topk1, gw1, part /* stage 0 */);
  proj_norm_kernel<<<dim3(B_ROWS / 128, E_EXP), 256, 0, stream>>>(
      x_h, D_IN, B_ROWS, Ur1t, Ub1, u1);
  proj_norm_kernel<<<dim3(N_EMB / 128, E_EXP), 256, 0, stream>>>(
      remb_h, D_IN, N_EMB, Vr1t, Vb1, v1);
  score_sample_kernel<<<B_ROWS / 16, 256, score_lds, stream>>>(
      u1, v1, topk1, gw1, rand_rea, sel1, out_rea_idx, out_rea_logp);

  // ---- concat xc = [x, rea_emb[sel]] (f32 + f16) ----
  concat_kernel<<<B_ROWS, 128, 0, stream>>>(x, rea_emb, sel1, xc, xc_h);

  // ---- stage 2: LLM selection ----
  gate_topk_kernel<<<B_ROWS / 256, 256, 0, stream>>>(
      xc, 2 * D_IN, g2w, g2b, topk2, gw2, part + 32 * 16 /* stage 1 */);
  proj_norm_kernel<<<dim3(B_ROWS / 128, E_EXP), 256, 0, stream>>>(
      xc_h, 2 * D_IN, B_ROWS, Ur2t, Ub2, u2);
  proj_norm_kernel<<<dim3(N_EMB / 128, E_EXP), 256, 0, stream>>>(
      lemb_h, D_IN, N_EMB, Vr2t, Vb2, v2);
  score_sample_kernel<<<B_ROWS / 16, 256, score_lds, stream>>>(
      u2, v2, topk2, gw2, rand_llm, sel2, out_llm_idx, out_llm_logp);

  // ---- aux losses ----
  aux_finalize_kernel<<<1, 1, 0, stream>>>(part, out_aux);
}
